// pretrain_autoencoder_LSTM_87522843558147
// MI455X (gfx1250) — compile-verified, matrix operand order checked by reading, not run on hardware
//
#include <hip/hip_runtime.h>
#include <hip/hip_bf16.h>

// Problem dims (match reference)
#define BB 128   // batch
#define SS 512   // seq
#define II 128   // input width
#define HH 512   // hidden
#define G4 2048  // 4*H gate width

typedef __attribute__((ext_vector_type(16))) __bf16 v16bf;
typedef __attribute__((ext_vector_type(8)))  float  v8f;
typedef __attribute__((ext_vector_type(8)))  unsigned short u16x8;
typedef __attribute__((ext_vector_type(16))) unsigned short u16x16;

__device__ __forceinline__ unsigned short f32_to_bf16_rne(float f) {
  unsigned u = __float_as_uint(f);
  if ((u & 0x7F800000u) == 0x7F800000u) return (unsigned short)(u >> 16); // inf/nan
  unsigned r = u + 0x7FFFu + ((u >> 16) & 1u);
  return (unsigned short)(r >> 16);
}

// Load a 16x32 bf16 fragment (A layout, ISA 7.12.2) from a row-major [rows x rowstride]
// bf16 matrix at (row0, k0). For C = A * W^T the B fragment is the identical load
// applied to W's rows (B[k][n] == W[n][k]), so one loader serves both operands.
// lane<16 : row=row0+lane,    Ks {k0..k0+7,  k0+16..k0+23}
// lane>=16: row=row0+lane-16, Ks {k0+8..+15, k0+24..+31}
__device__ __forceinline__ v16bf load_frag_bf16(const unsigned short* __restrict__ base,
                                                int rowstride, int row0, int k0) {
  int lane = threadIdx.x & 31;
  const unsigned short* p = base + (row0 + (lane & 15)) * rowstride + (k0 + ((lane >> 4) << 3));
  u16x8 lo = *(const u16x8*)p;          // 16B load
  u16x8 hi = *(const u16x8*)(p + 16);   // 16B load
  u16x16 full = __builtin_shufflevector(lo, hi, 0,1,2,3,4,5,6,7,8,9,10,11,12,13,14,15);
  union { u16x16 u; v16bf v; } cvt; cvt.u = full;
  return cvt.v;
}

#define WMMA_BF16(ACC, A, Bm) \
  __builtin_amdgcn_wmma_f32_16x16x32_bf16(false, (A), false, (Bm), (short)0, (ACC), false, false)

// ---------------- prep kernels ----------------
__global__ void k_f32_to_bf16(const float* __restrict__ src, unsigned short* __restrict__ dst, int n) {
  int i = blockIdx.x * 256 + threadIdx.x;
  if (i < n) dst[i] = f32_to_bf16_rne(src[i]);
}

__global__ void k_bias_sum(const float* __restrict__ a, const float* __restrict__ b,
                           float* __restrict__ o, int n) {
  int i = blockIdx.x * 256 + threadIdx.x;
  if (i < n) o[i] = a[i] + b[i];
}

// ---------------- fused LSTM cell step ----------------
// gates[B=128, 4H=2048] = Ax[128,K0]*Wih^T + Ah[128,H]*Whh^T + bias; then cell update.
// grid: 128 blocks = 4 row-stripes (32 rows) x 32 h-col-tiles; block: 128 thr = 4 waves,
// wave w = gate w. Each wave keeps TWO accumulators (rows r0..r0+15, r0+16..r0+31) so the
// weight (B) fragment loaded from L2 is reused twice -> halves per-step weight traffic.
__global__ __launch_bounds__(128)
void k_lstm_cell(const unsigned short* __restrict__ Ax, int ax_stride, int K0,
                 const unsigned short* __restrict__ Wih,   // [2048 x K0] bf16 row-major
                 const unsigned short* __restrict__ Ah,    // [128 x H]  bf16 (h_{t-1})
                 const unsigned short* __restrict__ Whh,   // [2048 x H] bf16
                 const float*          __restrict__ bias,  // [2048] (b_ih + b_hh)
                 float*                __restrict__ c_st,  // [128 x H] fp32, in-place
                 unsigned short*       __restrict__ h_out) // [128 x H] bf16 (h_t)
{
  __shared__ float gbuf[4][32][17];  // +1 pad: avoid LDS bank conflicts
  const int wave = threadIdx.x >> 5;
  const int lane = threadIdx.x & 31;
  const int rt = blockIdx.x >> 5;          // 0..3   batch stripe (32 rows)
  const int ht = blockIdx.x & 31;          // 0..31  hidden-col tile
  const int row0  = rt * 32;
  const int wrow0 = wave * HH + ht * 16;   // rows of W == cols of gates for this gate

  v8f acc0 = {}, acc1 = {};
  for (int k = 0; k < K0; k += 32) {       // x-projection
    v16bf b  = load_frag_bf16(Wih, K0,        wrow0,     k);
    v16bf a0 = load_frag_bf16(Ax,  ax_stride, row0,      k);
    v16bf a1 = load_frag_bf16(Ax,  ax_stride, row0 + 16, k);
    acc0 = WMMA_BF16(acc0, a0, b);
    acc1 = WMMA_BF16(acc1, a1, b);
  }
  #pragma unroll 4
  for (int k = 0; k < HH; k += 32) {       // recurrent projection
    v16bf b  = load_frag_bf16(Whh, HH, wrow0,     k);
    v16bf a0 = load_frag_bf16(Ah,  HH, row0,      k);
    v16bf a1 = load_frag_bf16(Ah,  HH, row0 + 16, k);
    acc0 = WMMA_BF16(acc0, a0, b);
    acc1 = WMMA_BF16(acc1, a1, b);
  }

  // C/D layout: lane<16 -> rows 0..7, lane>=16 -> rows 8..15; col = lane&15
  const int col   = lane & 15;
  const int rbase = (lane >> 4) * 8;
  const float bv  = bias[wrow0 + col];
  #pragma unroll
  for (int g = 0; g < 8; ++g) {
    gbuf[wave][rbase + g][col]      = acc0[g] + bv;
    gbuf[wave][16 + rbase + g][col] = acc1[g] + bv;
  }
  __syncthreads();

  // pointwise cell update: 512 stripe elements over 128 threads
  for (int idx = threadIdx.x; idx < 512; idx += 128) {
    int r = idx >> 4, cc = idx & 15;
    float gi = gbuf[0][r][cc], gf = gbuf[1][r][cc];
    float gg = gbuf[2][r][cc], go = gbuf[3][r][cc];
    int b_   = row0 + r;
    int hcol = ht * 16 + cc;
    float cold = c_st[b_ * HH + hcol];
    float si = 1.f / (1.f + __expf(-gi));
    float sf = 1.f / (1.f + __expf(-gf));
    float so = 1.f / (1.f + __expf(-go));
    float cn = sf * cold + si * tanhf(gg);
    c_st[b_ * HH + hcol]  = cn;
    h_out[b_ * HH + hcol] = f32_to_bf16_rne(so * tanhf(cn));
  }
}

// ---------------- decoder linear feedback: pred = h1 * Wlin^T ----------------
// grid: 32 blocks = 4 row-stripes x 8 col-tiles; 1 wave per block, 2 accs (B reuse).
__global__ __launch_bounds__(32)
void k_linear(const unsigned short* __restrict__ h1,    // [128 x H] bf16
              const unsigned short* __restrict__ Wlin,  // [I x H]   bf16
              float* __restrict__ out,                  // [B,S,I] fp32
              unsigned short* __restrict__ pred_bf,     // [B x I]   bf16 feedback
              int t)
{
  const int rt = blockIdx.x >> 3;          // 0..3  (32-row stripe)
  const int ct = blockIdx.x & 7;           // 0..7
  const int lane = threadIdx.x;
  const int row0 = rt * 32;
  v8f acc0 = {}, acc1 = {};
  #pragma unroll 4
  for (int k = 0; k < HH; k += 32) {
    v16bf b  = load_frag_bf16(Wlin, HH, ct * 16,   k);
    v16bf a0 = load_frag_bf16(h1,   HH, row0,      k);
    v16bf a1 = load_frag_bf16(h1,   HH, row0 + 16, k);
    acc0 = WMMA_BF16(acc0, a0, b);
    acc1 = WMMA_BF16(acc1, a1, b);
  }
  const int col = lane & 15, rbase = (lane >> 4) * 8;
  #pragma unroll
  for (int g = 0; g < 8; ++g) {
    int i_  = ct * 16 + col;
    int b0_ = row0 + rbase + g;
    int b1_ = row0 + 16 + rbase + g;
    float v0 = acc0[g], v1 = acc1[g];
    out[(size_t)b0_ * SS * II + (size_t)t * II + i_] = v0;
    out[(size_t)b1_ * SS * II + (size_t)t * II + i_] = v1;
    pred_bf[b0_ * II + i_] = f32_to_bf16_rne(v0);
    pred_bf[b1_ * II + i_] = f32_to_bf16_rne(v1);
  }
}

// ---------------- host ----------------
extern "C" void kernel_launch(void* const* d_in, const int* in_sizes, int n_in,
                              void* d_out, int out_size, void* d_ws, size_t ws_size,
                              hipStream_t stream) {
  (void)in_sizes; (void)n_in; (void)out_size; (void)ws_size;
  const float* x = (const float*)d_in[0];
  // per layer: W_ih, W_hh, b_ih, b_hh at indices 1+4*L .. 4+4*L ; L = en0,en1,de0,de1
  const float* Wih_f[4], * Whh_f[4], * bih_f[4], * bhh_f[4];
  for (int L = 0; L < 4; ++L) {
    Wih_f[L] = (const float*)d_in[1 + 4 * L];
    Whh_f[L] = (const float*)d_in[2 + 4 * L];
    bih_f[L] = (const float*)d_in[3 + 4 * L];
    bhh_f[L] = (const float*)d_in[4 + 4 * L];
  }
  const float* Wlin_f = (const float*)d_in[17];
  const int Kin[4] = { II, HH, II, HH };   // W_ih inner dims per layer

  // --- workspace bump allocator (256B aligned) ---
  size_t off = 0;
  auto alloc = [&](size_t bytes) -> void* {
    void* p = (char*)d_ws + off;
    off += (bytes + 255) & ~(size_t)255;
    return p;
  };
  unsigned short* x_bf = (unsigned short*)alloc((size_t)BB * SS * II * 2);
  unsigned short *Wih_bf[4], *Whh_bf[4];
  float* bias[4];
  for (int L = 0; L < 4; ++L) {
    Wih_bf[L] = (unsigned short*)alloc((size_t)G4 * Kin[L] * 2);
    Whh_bf[L] = (unsigned short*)alloc((size_t)G4 * HH * 2);
    bias[L]   = (float*)alloc((size_t)G4 * 4);
  }
  unsigned short* Wlin_bf = (unsigned short*)alloc((size_t)II * HH * 2);
  unsigned short* h0buf[2], * h1buf[2];
  for (int p = 0; p < 2; ++p) {
    h0buf[p] = (unsigned short*)alloc((size_t)BB * HH * 2);
    h1buf[p] = (unsigned short*)alloc((size_t)BB * HH * 2);
  }
  float* c0 = (float*)alloc((size_t)BB * HH * 4);
  float* c1 = (float*)alloc((size_t)BB * HH * 4);
  unsigned short* pred_bf = (unsigned short*)alloc((size_t)BB * II * 2);

  auto cvt = [&](const float* s, unsigned short* d, int n) {
    k_f32_to_bf16<<<dim3((n + 255) / 256), dim3(256), 0, stream>>>(s, d, n);
  };

  // --- prep: bf16 weight/x images + fused biases + zeroed state (every call) ---
  cvt(x, x_bf, BB * SS * II);
  for (int L = 0; L < 4; ++L) {
    cvt(Wih_f[L], Wih_bf[L], G4 * Kin[L]);
    cvt(Whh_f[L], Whh_bf[L], G4 * HH);
    k_bias_sum<<<dim3((G4 + 255) / 256), dim3(256), 0, stream>>>(bih_f[L], bhh_f[L], bias[L], G4);
  }
  cvt(Wlin_f, Wlin_bf, II * HH);
  for (int p = 0; p < 2; ++p) {
    hipMemsetAsync(h0buf[p], 0, (size_t)BB * HH * 2, stream);
    hipMemsetAsync(h1buf[p], 0, (size_t)BB * HH * 2, stream);
  }
  hipMemsetAsync(c0, 0, (size_t)BB * HH * 4, stream);
  hipMemsetAsync(c1, 0, (size_t)BB * HH * 4, stream);
  hipMemsetAsync(pred_bf, 0, (size_t)BB * II * 2, stream);

  const dim3 cg(128), cb(128);

  // --- encoder: h read index = t&1, write index = (t&1)^1 (race-free ping-pong) ---
  for (int t = 0; t < SS; ++t) {
    int p = t & 1;
    k_lstm_cell<<<cg, cb, 0, stream>>>(x_bf + (size_t)t * II, SS * II, II,
                                       Wih_bf[0], h0buf[p], Whh_bf[0], bias[0],
                                       c0, h0buf[p ^ 1]);
    k_lstm_cell<<<cg, cb, 0, stream>>>(h0buf[p ^ 1], HH, HH,
                                       Wih_bf[1], h1buf[p], Whh_bf[1], bias[1],
                                       c1, h1buf[p ^ 1]);
  }
  // after t=511 (odd), current h lives in index 0 -> decoder parity formula matches.

  // --- decoder: closed loop through pred_bf ---
  for (int t = 0; t < SS; ++t) {
    int p = t & 1;
    k_lstm_cell<<<cg, cb, 0, stream>>>(pred_bf, II, II,
                                       Wih_bf[2], h0buf[p], Whh_bf[2], bias[2],
                                       c0, h0buf[p ^ 1]);
    k_lstm_cell<<<cg, cb, 0, stream>>>(h0buf[p ^ 1], HH, HH,
                                       Wih_bf[3], h1buf[p], Whh_bf[3], bias[3],
                                       c1, h1buf[p ^ 1]);
    k_linear<<<dim3(32), dim3(32), 0, stream>>>(h1buf[p ^ 1], Wlin_bf,
                                                (float*)d_out, pred_bf, t);
  }
}
